// KGPrompt_71390946394612
// MI455X (gfx1250) — matrix operand order repaired and checked by reading, not compiled, not run on hardware
//
#include <hip/hip_runtime.h>
#include <hip/hip_bf16.h>

// ---------------------------------------------------------------------------
// KGPrompt forward for MI455X (gfx1250, wave32, WMMA).
//  * Entity branch computed only for the 512 gathered (b,l) rows.
//  * RGCN combine expressed as one GEMM: Am[512,6528] x Wm[6528,384].
//  * All GEMMs via a shared bf16 WMMA kernel (v_wmma_f32_16x16x32_bf16),
//    64x64x32 block tile, 8 waves, fused bias/ReLU/residual epilogue.
//    B tile is stored TRANSPOSED in LDS so every fragment load is a
//    contiguous ds_load_b128 (no scalar u16 gathers in the inner loop).
//  * Final pp2 GEMM epilogue scatters directly into the transposed
//    [12,2,16,12,256,64] output layout (no extra 600MB transpose pass).
// ---------------------------------------------------------------------------

typedef __attribute__((ext_vector_type(16))) __bf16 v16bf;
typedef __attribute__((ext_vector_type(8)))  __bf16 v8bf;
typedef __attribute__((ext_vector_type(4)))  __bf16 v4bf;
typedef __attribute__((ext_vector_type(8)))  float  v8f;
typedef __attribute__((ext_vector_type(4)))  float  v4f;

#define N_ENTITY 25000
#define D_DIM 384
#define H_DIM 768
#define N_REL 16
#define N_BASES 8
#define N_EDGES 400000
#define BATCH 16
#define ENT_LEN 32
#define TOK_LEN 256
#define N_ROWS (BATCH * ENT_LEN)          // 512 entity rows
#define T_ROWS (BATCH * TOK_LEN)          // 4096 token rows
#define KCOMB ((1 + N_REL) * D_DIM)       // 6528 = x | agg_mean per relation
#define PP2_N (12 * 2 * H_DIM)            // 18432
#define SLOT_EMPTY 0x7fffffff

// ------------------------------- utility kernels ---------------------------

__global__ void k_fill_i32(int* p, int v, int n) {
  int i = blockIdx.x * blockDim.x + threadIdx.x;
  if (i < n) p[i] = v;
}

__global__ void k_zero_u32(unsigned* p, int n) {
  int i = blockIdx.x * blockDim.x + threadIdx.x;
  if (i < n) p[i] = 0u;
}

__global__ void k_mark_slot(const int* __restrict__ entity_ids, int* slot, int n) {
  int i = blockIdx.x * blockDim.x + threadIdx.x;
  if (i < n) atomicMin(&slot[entity_ids[i]], i);
}

// Scatter x[src] into agg[slot(dst)*16 + rel]; one wave per edge iteration.
__global__ void k_edge_scatter(const int* __restrict__ src, const int* __restrict__ dst,
                               const int* __restrict__ etype, const int* __restrict__ slot,
                               const float* __restrict__ x, float* __restrict__ agg,
                               int* __restrict__ cnt, int nE) {
  int wid = (blockIdx.x * blockDim.x + threadIdx.x) >> 5;
  int lane = threadIdx.x & 31;
  int nW = (gridDim.x * blockDim.x) >> 5;
  for (int e = wid; e < nE; e += nW) {
    int d = dst[e];
    int s = slot[d];
    if (s == SLOT_EMPTY) continue;
    int r = etype[e];
    int sr = src[e];
    if (lane == 0) atomicAdd(&cnt[s * N_REL + r], 1);
    const float* xs = x + (size_t)sr * D_DIM;
    float* ag = agg + ((size_t)s * N_REL + r) * D_DIM;
    for (int j = lane; j < D_DIM; j += 32) atomicAdd(&ag[j], xs[j]);
  }
}

// Wm[k][j]: rows 0..383 = root; rows 384*(1+r)+d = w_rel[r][d][j] = sum_b comp[r,b]*basis[b,d,j]
__global__ void k_build_Wm(const float* __restrict__ root, const float* __restrict__ comp,
                           const float* __restrict__ basis, float* __restrict__ Wm) {
  int idx = blockIdx.x * blockDim.x + threadIdx.x;
  const int n = KCOMB * D_DIM;
  if (idx >= n) return;
  int k = idx / D_DIM;
  int j = idx - k * D_DIM;
  float v;
  if (k < D_DIM) {
    v = root[k * D_DIM + j];
  } else {
    int kk = k - D_DIM;
    int r = kk / D_DIM;
    int d = kk - r * D_DIM;
    v = 0.f;
#pragma unroll
    for (int b = 0; b < N_BASES; ++b)
      v += comp[r * N_BASES + b] * basis[((size_t)b * D_DIM + d) * D_DIM + j];
  }
  Wm[idx] = v;
}

// Am[i][c]: c<384 -> node_embeds[entity_ids[i]][c]; else agg_mean[slot][r][j]
__global__ void k_build_Am(const int* __restrict__ entity_ids, const int* __restrict__ slot,
                           const float* __restrict__ node_embeds, const float* __restrict__ agg,
                           const int* __restrict__ cnt, float* __restrict__ Am) {
  int idx = blockIdx.x * blockDim.x + threadIdx.x;
  const int n = N_ROWS * KCOMB;
  if (idx >= n) return;
  int i = idx / KCOMB;
  int c = idx - i * KCOMB;
  int e = entity_ids[i];
  float v;
  if (c < D_DIM) {
    v = node_embeds[(size_t)e * D_DIM + c];
  } else {
    int cc = c - D_DIM;
    int r = cc / D_DIM;
    int j = cc - r * D_DIM;
    int s = slot[e];
    int cn = cnt[s * N_REL + r];
    float denom = (float)(cn < 1 ? 1 : cn);
    v = agg[((size_t)s * N_REL + r) * D_DIM + j] / denom;
  }
  Am[idx] = v;
}

// ------------------------------- WMMA GEMM ---------------------------------
// C[M,N] = A[M,K] x B[K,N] (+bias)(ReLU)(+res). M%64==0, N%64==0, K%32==0.
// Block: 256 threads = 8 waves. Block tile 64x64, K step 32.
// Waves 0-3 -> col tiles {0,1}, waves 4-7 -> {2,3}; row tile = wave&3.
__global__ __launch_bounds__(256) void gemm_bf16_kernel(
    const float* __restrict__ A, const float* __restrict__ B,
    const float* __restrict__ bias, const float* __restrict__ res,
    float* __restrict__ C, int M, int N, int K, int res_ld, int relu, int scatter) {
  __shared__ alignas(32) __bf16 As[64 * 32];  // [row][k]
  __shared__ alignas(32) __bf16 Bt[64 * 32];  // [col][k]  (transposed tile)
  const int tid = threadIdx.x;
  const int row0 = blockIdx.y * 64;
  const int col0 = blockIdx.x * 64;
  const int wave = tid >> 5, lane = tid & 31;
  const int half = lane >> 4, mn = lane & 15;
  const int wr = wave & 3;
  const int wc = (wave >> 2) * 2;

  v8f acc0 = {}, acc1 = {};
  for (int kb = 0; kb < K; kb += 32) {
    // ---- stage A tile: 64x32, float4 global loads, contiguous bf16 stores
#pragma unroll
    for (int i = 0; i < 2; ++i) {
      int idx4 = i * 256 + tid;          // 512 float4 = 2048 floats
      int r = idx4 >> 3;                 // row 0..63
      int c = (idx4 & 7) * 4;            // k 0,4,..28
      v4f f = *(const v4f*)&A[(size_t)(row0 + r) * K + kb + c];
      v4bf h;
#pragma unroll
      for (int u = 0; u < 4; ++u) h[u] = (__bf16)f[u];
      *(v4bf*)&As[r * 32 + c] = h;
    }
    // ---- stage B tile: 32x64 read coalesced, stored transposed [col][k]
#pragma unroll
    for (int i = 0; i < 2; ++i) {
      int idx4 = i * 256 + tid;
      int r = idx4 >> 4;                 // k row 0..31
      int c = (idx4 & 15) * 4;           // col 0,4,..60
      v4f f = *(const v4f*)&B[(size_t)(kb + r) * N + col0 + c];
#pragma unroll
      for (int u = 0; u < 4; ++u) Bt[(c + u) * 32 + r] = (__bf16)f[u];
    }
    if (kb + 32 < K) {                   // gfx1250 global_prefetch_b8 next tile
      __builtin_prefetch(&A[(size_t)(row0 + (tid >> 3)) * K + kb + 32], 0, 1);
      __builtin_prefetch(&B[(size_t)(kb + 32 + (tid >> 4)) * N + col0], 0, 1);
    }
    __syncthreads();

    // A frag: lane<16 holds K {0..15}, lane>=16 holds {16..31};
    // both halves are two contiguous 16B runs -> ds_load_b128 pairs.
    const __bf16* arow = &As[(wr * 16 + mn) * 32];
    v8bf alo = *(const v8bf*)&arow[half * 16];
    v8bf ahi = *(const v8bf*)&arow[half * 16 + 8];
    v16bf af = __builtin_shufflevector(alo, ahi, 0, 1, 2, 3, 4, 5, 6, 7,
                                       8, 9, 10, 11, 12, 13, 14, 15);
    // B frag: col fixed per lane, K = half*16 .. half*16+15 contiguous.
    v16bf bf0 = *(const v16bf*)&Bt[(wc * 16 + mn) * 32 + half * 16];
    v16bf bf1 = *(const v16bf*)&Bt[((wc + 1) * 16 + mn) * 32 + half * 16];

    acc0 = __builtin_amdgcn_wmma_f32_16x16x32_bf16(false, af, false, bf0,
                                                   (short)0, acc0, false, false);
    acc1 = __builtin_amdgcn_wmma_f32_16x16x32_bf16(false, af, false, bf1,
                                                   (short)0, acc1, false, false);
    __syncthreads();
  }

  // Epilogue. C/D layout: VGPR j -> row (j + 8*half), col (lane&15).
#pragma unroll
  for (int j = 0; j < 8; ++j) {
    int grow = row0 + wr * 16 + j + half * 8;
#pragma unroll
    for (int t = 0; t < 2; ++t) {
      int gcol = col0 + (wc + t) * 16 + mn;
      float v = (t == 0) ? acc0[j] : acc1[j];
      if (bias) v += bias[gcol];
      if (relu) v = fmaxf(v, 0.f);
      if (res) v += res[(size_t)grow * res_ld + gcol];
      if (scatter) {
        // grow = b*256+t ; gcol = lb*768 + h*64 + hd ; out[lb,b,h,t,hd]
        int bb = grow >> 8, tt = grow & 255;
        int lb = gcol / H_DIM;
        int rem = gcol - lb * H_DIM;
        int h = rem >> 6, hd = rem & 63;
        size_t oi = ((((size_t)lb * BATCH + bb) * 12 + h) * TOK_LEN + tt) * 64 + hd;
        C[oi] = v;
      } else {
        C[(size_t)grow * N + gcol] = v;
      }
    }
  }
}

// ------------------------------ cross attention ----------------------------
// One wave per (b,t) row; 32 entities == 32 lanes for the softmax.
__global__ __launch_bounds__(256) void k_attention(
    const float* __restrict__ q, const float* __restrict__ tok,
    const float* __restrict__ entH, float* __restrict__ out) {
  int row = blockIdx.x * (blockDim.x >> 5) + (threadIdx.x >> 5);  // 0..4095
  int lane = threadIdx.x & 31;
  int b = row >> 8;
  const float* qr = q + (size_t)row * H_DIM;
  const float* er = entH + (size_t)b * ENT_LEN * H_DIM;

  float ql[24];
#pragma unroll
  for (int i = 0; i < 24; ++i) ql[i] = qr[lane + 32 * i];

  float s = 0.f;
  for (int e = 0; e < 32; ++e) {
    const float* ee = er + e * H_DIM;
    float d = 0.f;
#pragma unroll
    for (int i = 0; i < 24; ++i) d += ql[i] * ee[lane + 32 * i];
    for (int off = 16; off; off >>= 1) d += __shfl_xor(d, off, 32);
    if (lane == e) s = d * (1.0f / (float)H_DIM);
  }
  float m = s;
  for (int off = 16; off; off >>= 1) m = fmaxf(m, __shfl_xor(m, off, 32));
  float p = expf(s - m);
  float sum = p;
  for (int off = 16; off; off >>= 1) sum += __shfl_xor(sum, off, 32);
  p /= sum;

  const float* tr = tok + (size_t)row * H_DIM;
  float* orow = out + (size_t)row * H_DIM;
#pragma unroll
  for (int i = 0; i < 24; ++i) {
    int k = lane + 32 * i;
    float acc = tr[k];
    for (int e = 0; e < 32; ++e) {
      float pe = __shfl(p, e, 32);
      acc += pe * er[e * H_DIM + k];
    }
    orow[k] = acc;
  }
}

// -------------------------------- host side --------------------------------

static void launch_gemm(const float* A, const float* B, const float* bias,
                        const float* res, int res_ld, float* C,
                        int M, int N, int K, int relu, int scatter, hipStream_t s) {
  dim3 g(N / 64, M / 64), b(256);
  gemm_bf16_kernel<<<g, b, 0, s>>>(A, B, bias, res, C, M, N, K, res_ld, relu, scatter);
}

extern "C" void kernel_launch(void* const* d_in, const int* in_sizes, int n_in,
                              void* d_out, int out_size, void* d_ws, size_t ws_size,
                              hipStream_t stream) {
  const float *tokenE, *nodeE, *basis, *comp, *root, *rgcn_bias;
  const float *ep1aw, *ep1ab, *ep1bw, *ep1bb, *ep2w, *ep2b;
  const float *tp1aw, *tp1ab, *tp1bw, *tp1bb, *tp2w, *tp2b, *crossw;
  const float *pp1aw, *pp1ab, *pp1bw, *pp1bb, *pp2w, *pp2b;
  const int *ent_ids, *edge_index, *edge_type;

#define F(i) ((const float*)d_in[(i)])
#define I(i) ((const int*)d_in[(i)])
  if (in_sizes[0] == BATCH * TOK_LEN * H_DIM) {
    // Insertion-order flattening of setup_inputs()
    tokenE = F(0);  nodeE = F(1);  basis = F(2);  comp = F(3);  root = F(4);
    rgcn_bias = F(5);
    ep1aw = F(6);  ep1ab = F(7);  ep1bw = F(8);  ep1bb = F(9);
    ep2w = F(10);  ep2b = F(11);
    tp1aw = F(12); tp1ab = F(13); tp1bw = F(14); tp1bb = F(15);
    tp2w = F(16);  tp2b = F(17);  crossw = F(18);
    pp1aw = F(19); pp1ab = F(20); pp1bw = F(21); pp1bb = F(22);
    pp2w = F(23);  pp2b = F(24);
    ent_ids = I(25); edge_index = I(26); edge_type = I(27);
  } else {
    // jax pytree flattening: dict keys sorted alphabetically at every level
    edge_index = I(0); edge_type = I(1); ent_ids = I(2);
    basis = F(3); comp = F(4); crossw = F(5);
    ep1ab = F(6);  ep1aw = F(7);  ep1bb = F(8);  ep1bw = F(9);
    ep2b = F(10);  ep2w = F(11);  nodeE = F(12);
    pp1ab = F(13); pp1aw = F(14); pp1bb = F(15); pp1bw = F(16);
    pp2b = F(17);  pp2w = F(18);  rgcn_bias = F(19); root = F(20);
    tp1ab = F(21); tp1aw = F(22); tp1bb = F(23); tp1bw = F(24);
    tp2b = F(25);  tp2w = F(26);  tokenE = F(27);
  }
#undef F
#undef I

  const int* e_src = edge_index;            // edge_index[0]
  const int* e_dst = edge_index + N_EDGES;  // edge_index[1]

  // ---- carve workspace ----
  char* ws = (char*)d_ws;
  size_t off = 0;
  auto carve = [&](size_t bytes) -> char* {
    char* p = ws + off;
    off += (bytes + 255) & ~(size_t)255;
    return p;
  };
  int*   slot   = (int*)carve((size_t)N_ENTITY * 4);
  int*   cnt    = (int*)carve((size_t)N_ROWS * N_REL * 4);
  float* agg    = (float*)carve((size_t)N_ROWS * N_REL * D_DIM * 4);
  float* Am     = (float*)carve((size_t)N_ROWS * KCOMB * 4);
  float* Wm     = (float*)carve((size_t)KCOMB * D_DIM * 4);
  float* ent0   = (float*)carve((size_t)N_ROWS * D_DIM * 4);
  float* h1e    = (float*)carve((size_t)N_ROWS * (D_DIM / 2) * 4);
  float* ent1   = (float*)carve((size_t)N_ROWS * D_DIM * 4);
  float* entH   = (float*)carve((size_t)N_ROWS * H_DIM * 4);
  float* th1    = (float*)carve((size_t)T_ROWS * (H_DIM / 2) * 4);  // also ph1
  float* tok1   = (float*)carve((size_t)T_ROWS * H_DIM * 4);        // also prompt1
  float* tokp   = (float*)carve((size_t)T_ROWS * H_DIM * 4);
  float* qbuf   = (float*)carve((size_t)T_ROWS * H_DIM * 4);
  float* prom0  = (float*)carve((size_t)T_ROWS * H_DIM * 4);
  (void)ws_size; (void)n_in; (void)out_size;

  float* out = (float*)d_out;

  // ---- RGCN front end (512 gathered rows only) ----
  k_fill_i32<<<(N_ENTITY + 255) / 256, 256, 0, stream>>>(slot, SLOT_EMPTY, N_ENTITY);
  k_mark_slot<<<(N_ROWS + 255) / 256, 256, 0, stream>>>(ent_ids, slot, N_ROWS);
  {
    int nAgg = N_ROWS * N_REL * D_DIM;
    k_zero_u32<<<(nAgg + 255) / 256, 256, 0, stream>>>((unsigned*)agg, nAgg);
    int nCnt = N_ROWS * N_REL;
    k_zero_u32<<<(nCnt + 255) / 256, 256, 0, stream>>>((unsigned*)cnt, nCnt);
  }
  k_edge_scatter<<<1024, 256, 0, stream>>>(e_src, e_dst, edge_type, slot,
                                           nodeE, agg, cnt, N_EDGES);
  {
    int nWm = KCOMB * D_DIM;
    k_build_Wm<<<(nWm + 255) / 256, 256, 0, stream>>>(root, comp, basis, Wm);
    int nAm = N_ROWS * KCOMB;
    k_build_Am<<<(nAm + 255) / 256, 256, 0, stream>>>(ent_ids, slot, nodeE, agg, cnt, Am);
  }
  // ent0 = Am x Wm + rgcn_bias + x (x = first 384 cols of Am)   [RGCN + residual]
  launch_gemm(Am, Wm, rgcn_bias, Am, KCOMB, ent0, N_ROWS, D_DIM, KCOMB, 0, 0, stream);

  // ---- entity projections ----
  launch_gemm(ent0, ep1aw, ep1ab, nullptr, 0, h1e, N_ROWS, D_DIM / 2, D_DIM, 1, 0, stream);
  launch_gemm(h1e, ep1bw, ep1bb, ent0, D_DIM, ent1, N_ROWS, D_DIM, D_DIM / 2, 0, 0, stream);
  launch_gemm(ent1, ep2w, ep2b, nullptr, 0, entH, N_ROWS, H_DIM, D_DIM, 0, 0, stream);

  // ---- token projections ----
  launch_gemm(tokenE, tp1aw, tp1ab, nullptr, 0, th1, T_ROWS, H_DIM / 2, H_DIM, 1, 0, stream);
  launch_gemm(th1, tp1bw, tp1bb, tokenE, H_DIM, tok1, T_ROWS, H_DIM, H_DIM / 2, 0, 0, stream);
  launch_gemm(tok1, tp2w, tp2b, nullptr, 0, tokp, T_ROWS, H_DIM, H_DIM, 0, 0, stream);

  // ---- cross attention ----
  launch_gemm(tokp, crossw, nullptr, nullptr, 0, qbuf, T_ROWS, H_DIM, H_DIM, 0, 0, stream);
  k_attention<<<T_ROWS / 8, 256, 0, stream>>>(qbuf, tokp, entH, prom0);

  // ---- prompt projections (reuse th1 as ph1, tok1 as prompt1) ----
  launch_gemm(prom0, pp1aw, pp1ab, nullptr, 0, th1, T_ROWS, H_DIM / 2, H_DIM, 1, 0, stream);
  launch_gemm(th1, pp1bw, pp1bb, prom0, H_DIM, tok1, T_ROWS, H_DIM, H_DIM / 2, 0, 0, stream);
  // final GEMM scatter-writes directly into [12,2,16,12,256,64] layout
  launch_gemm(tok1, pp2w, pp2b, nullptr, 0, out, T_ROWS, PP2_N, H_DIM, 0, 1, stream);
}
